// BertEncoder_58188216926927
// MI455X (gfx1250) — matrix operand order repaired and checked by reading, not compiled, not run on hardware
//
#include <hip/hip_runtime.h>
#include <hip/hip_bf16.h>

typedef __attribute__((ext_vector_type(16))) __bf16 bf16x16;
typedef __attribute__((ext_vector_type(8)))  float  floatx8;

union Frag16 { uint4 u[2]; bf16x16 v; };

static __device__ __forceinline__ floatx8 wmma_bf16f32(bf16x16 a, bf16x16 b, floatx8 c) {
  // D = A(16x32 bf16) * B(32x16 bf16) + C(16x16 f32)
  return __builtin_amdgcn_wmma_f32_16x16x32_bf16(false, a, false, b, (short)0, c, false, false);
}

// ---------------- problem dims ----------------
#define N_B   8
#define N_S   1024
#define N_D   512
#define N_H   8
#define N_DK  64
#define N_DF  2048
#define N_L   6
#define N_S1  1025            // S + AGG token
#define N_SP  1056            // padded seq (multiple of 32)
#define N_MP  (N_B * N_SP)    // 8448 padded rows (33 * 256)

// =====================================================================
// Embedding: mask/random/shuffle + pos-emb + LayerNorm, AGG row, padding.
// Also writes the `emb` passthrough and `mask_pos` outputs.
// =====================================================================
__global__ __launch_bounds__(256) void embed_kernel(
    const float* __restrict__ emb, const float* __restrict__ rnd,
    const int* __restrict__ perm, const float* __restrict__ pos,
    const float* __restrict__ pg, const float* __restrict__ pb,
    const float* __restrict__ mtok, const float* __restrict__ atok,
    float* __restrict__ x, __bf16* __restrict__ xb,
    float* __restrict__ out_emb, float* __restrict__ out_maskpos)
{
  const int row = blockIdx.x;           // b*SP + s
  const int b = row / N_SP;
  const int s = row % N_SP;
  const int tid = threadIdx.x;          // 256 threads, 2 elems each
  const size_t xo = (size_t)row * N_D;

  if (s == 0) {                         // AGG token (not layer-normed)
    #pragma unroll
    for (int e = 0; e < 2; ++e) {
      int d = tid + e * 256;
      float v = atok[d];
      x[xo + d] = v; xb[xo + d] = (__bf16)v;
    }
    return;
  }
  if (s >= N_S1) {                      // padded rows -> zero
    #pragma unroll
    for (int e = 0; e < 2; ++e) {
      int d = tid + e * 256;
      x[xo + d] = 0.0f; xb[xo + d] = (__bf16)0.0f;
    }
    return;
  }

  const int i = s - 1;
  const int tok = b * N_S + i;
  const float r0 = rnd[tok * 3 + 0], r1 = rnd[tok * 3 + 1], r2 = rnd[tok * 3 + 2];
  const bool ism = (r0 <= 0.2f);
  const float mA = (ism && r1 <= 0.8f) ? 1.0f : 0.0f;
  const float mR = (ism && r1 > 0.8f && r2 <= 0.5f) ? 1.0f : 0.0f;
  const float mN = (ism && r1 > 0.8f && r2 > 0.5f) ? 1.0f : 0.0f;
  const int pidx = perm[tok];
  const float coef = 1.0f - mA - mR;

  __shared__ float red[2][8];
  float vals[2], sum = 0.0f, sq = 0.0f;
  #pragma unroll
  for (int e = 0; e < 2; ++e) {
    int d = tid + e * 256;
    float ev = emb[(size_t)tok * N_D + d];
    float sv = emb[(size_t)pidx * N_D + d];
    float mv = ev * coef + mtok[d] * mA + sv * mR + pos[(size_t)i * N_D + d];
    vals[e] = mv; sum += mv; sq += mv * mv;
    out_emb[(size_t)tok * N_D + d] = ev;
  }
  #pragma unroll
  for (int off = 1; off < 32; off <<= 1) {
    sum += __shfl_xor(sum, off, 32);
    sq  += __shfl_xor(sq,  off, 32);
  }
  int lane = tid & 31, wid = tid >> 5;
  if (lane == 0) { red[0][wid] = sum; red[1][wid] = sq; }
  __syncthreads();
  float ts = 0.0f, tq = 0.0f;
  #pragma unroll
  for (int w = 0; w < 8; ++w) { ts += red[0][w]; tq += red[1][w]; }
  float mean = ts * (1.0f / N_D);
  float var  = tq * (1.0f / N_D) - mean * mean;
  float rstd = rsqrtf(var + 1e-6f);
  #pragma unroll
  for (int e = 0; e < 2; ++e) {
    int d = tid + e * 256;
    float o = (vals[e] - mean) * rstd * pg[d] + pb[d];
    x[xo + d] = o; xb[xo + d] = (__bf16)o;
  }
  if (tid == 0) out_maskpos[tok] = mA + mR + mN;
}

// attention bias per (b, padded key position)
__global__ void biasattn_kernel(const int* __restrict__ mask, float* __restrict__ bias) {
  int idx = blockIdx.x * 256 + threadIdx.x;
  if (idx >= N_B * N_SP) return;
  int b = idx / N_SP, s = idx % N_SP;
  float v;
  if (s == 0) v = 0.0f;
  else if (s <= N_S) v = (1.0f - (float)mask[b * N_S + s - 1]) * -1e9f;
  else v = -1e9f;  // padded keys
  bias[idx] = v;
}

// =====================================================================
// Weight conversion: fp32 [L,K,N] -> bf16 TRANSPOSED [L,N,K]
// (contiguous-K rows make both GEMM LDS tiles async-copy friendly)
// =====================================================================
__global__ void cvt_t_kernel(const float* __restrict__ in, __bf16* __restrict__ out,
                             int K, int N, long total) {
  long idx = (long)blockIdx.x * 256 + threadIdx.x;
  if (idx >= total) return;
  int k = (int)(idx % K);
  long r = idx / K;
  int n = (int)(r % N);
  int l = (int)(r / N);
  out[idx] = (__bf16)in[((size_t)l * K + k) * N + n];
}

// fused QKV weights: -> bf16 [L][1536 n][512 k]
__global__ void packqkv_kernel(const float* __restrict__ wq, const float* __restrict__ wk,
                               const float* __restrict__ wv, __bf16* __restrict__ out) {
  long idx = (long)blockIdx.x * 256 + threadIdx.x;
  if (idx >= (long)N_L * 1536 * 512) return;
  int k = (int)(idx % 512);
  long r = idx / 512;
  int n = (int)(r % 1536);
  int l = (int)(r / 1536);
  size_t base = ((size_t)l * 512 + k) * 512;
  float v = (n < 512) ? wq[base + n] : (n < 1024) ? wk[base + n - 512] : wv[base + n - 1024];
  out[idx] = (__bf16)v;
}

__global__ void packbqkv_kernel(const float* __restrict__ bq, const float* __restrict__ bk,
                                const float* __restrict__ bv, float* __restrict__ out) {
  int idx = blockIdx.x * 256 + threadIdx.x;
  if (idx >= N_L * 1536) return;
  int n = idx % 1536, l = idx / 1536;
  out[idx] = (n < 512) ? bq[l * 512 + n] : (n < 1024) ? bk[l * 512 + n - 512] : bv[l * 512 + n - 1024];
}

// =====================================================================
// bf16 WMMA GEMM: C[M,N] = A[M,K] * Wt[N,K]^T + bias (+GELU).
// Block 256 thr = 8 wave32; tile 256x128x32; wave tile 64x64 (4x4 wmma,
// ~32 FLOP per LDS byte). Double-buffered LDS tiles filled with
// GLOBAL_LOAD_ASYNC_TO_LDS_B128 (ASYNCcnt), overlapping the next tile
// fetch with WMMA on the current one.
// =====================================================================
#define GLDS 40  // padded LDS row stride in halfs (80B, 16B aligned)

static __device__ __forceinline__ float gelu_tanh(float v) {
  // branchless tanh-approx GELU (tanh via exp, maps to v_exp_f32)
  float u = 0.7978845608028654f * (v + 0.044715f * v * v * v);
  float e = __expf(-2.0f * fabsf(u));
  float t = (1.0f - e) / (1.0f + e);
  t = copysignf(t, u);
  return 0.5f * v * (1.0f + t);
}

__global__ __launch_bounds__(256) void gemm_kernel(
    const __bf16* __restrict__ A, const __bf16* __restrict__ Wt,
    const float* __restrict__ bias, float* __restrict__ outF,
    __bf16* __restrict__ outB, int M, int N, int K, int do_gelu)
{
  // per buffer: A 256*80B = 20480B, B 128*80B = 10240B -> 2*30720B = 60KB
  __shared__ __align__(16) __bf16 smem[2 * (256 + 128) * GLDS];

  const int tid = threadIdx.x;
  const int lane = tid & 31, wid = tid >> 5;
  const int waveM = wid >> 1, waveN = wid & 1;   // 4 x 2 wave grid
  const int bm = blockIdx.y * 256, bn = blockIdx.x * 128;
  const unsigned lbase = (unsigned)(uintptr_t)(&smem[0]);

  floatx8 acc[4][4];
  #pragma unroll
  for (int mt = 0; mt < 4; ++mt)
    #pragma unroll
    for (int nt = 0; nt < 4; ++nt)
      #pragma unroll
      for (int i = 0; i < 8; ++i) acc[mt][nt][i] = 0.0f;

  const int lnn = lane & 15, lhalf = lane >> 4, kb = lhalf * 8;

  // async-stage one 256x32 A tile + 128x32 W tile into buffer pbuf
  auto stage = [&](int k0, int pbuf) {
    unsigned baseA = lbase + (unsigned)pbuf * 30720u;
    unsigned baseB = baseA + 20480u;
    #pragma unroll
    for (int j = 0; j < 4; ++j) {       // A: 1024 chunks of 16B
      int c = tid * 4 + j;
      int row = c >> 2, sub = c & 3;
      const __bf16* ga = A + (size_t)(bm + row) * K + k0 + sub * 8;
      unsigned la = baseA + (unsigned)(row * 80 + sub * 16);
      asm volatile("global_load_async_to_lds_b128 %0, %1, off"
                   :: "v"(la), "v"((unsigned long long)(uintptr_t)ga) : "memory");
    }
    #pragma unroll
    for (int j = 0; j < 2; ++j) {       // B: 512 chunks of 16B
      int c = tid * 2 + j;
      int row = c >> 2, sub = c & 3;
      const __bf16* gw = Wt + (size_t)(bn + row) * K + k0 + sub * 8;
      unsigned lb = baseB + (unsigned)(row * 80 + sub * 16);
      asm volatile("global_load_async_to_lds_b128 %0, %1, off"
                   :: "v"(lb), "v"((unsigned long long)(uintptr_t)gw) : "memory");
    }
  };

  const int nk = K >> 5;
  stage(0, 0);
  int p = 0;
  for (int t = 0; t < nk; ++t) {
    asm volatile("s_wait_asynccnt 0" ::: "memory");
    __syncthreads();                       // tile t resident in buffer p
    if (t + 1 < nk) stage((t + 1) << 5, p ^ 1);

    const int hA = p * 15360;              // half offsets inside smem
    const int hB = hA + 10240;
    Frag16 afr[4], bfr[4];
    #pragma unroll
    for (int mt = 0; mt < 4; ++mt) {
      int r = waveM * 64 + mt * 16 + lnn;
      afr[mt].u[0] = *(const uint4*)(&smem[hA + r * GLDS + kb]);
      afr[mt].u[1] = *(const uint4*)(&smem[hA + r * GLDS + 16 + kb]);
    }
    #pragma unroll
    for (int nt = 0; nt < 4; ++nt) {
      int c = waveN * 64 + nt * 16 + lnn;
      bfr[nt].u[0] = *(const uint4*)(&smem[hB + c * GLDS + kb]);
      bfr[nt].u[1] = *(const uint4*)(&smem[hB + c * GLDS + 16 + kb]);
    }
    #pragma unroll
    for (int mt = 0; mt < 4; ++mt)
      #pragma unroll
      for (int nt = 0; nt < 4; ++nt)
        acc[mt][nt] = wmma_bf16f32(afr[mt].v, bfr[nt].v, acc[mt][nt]);
    p ^= 1;
  }

  // epilogue: D layout — lane n = lane&15, VGPR v -> m = v + 8*(lane>=16)
  #pragma unroll
  for (int nt = 0; nt < 4; ++nt) {
    int gn = bn + waveN * 64 + nt * 16 + lnn;
    float bz = bias[gn];
    #pragma unroll
    for (int mt = 0; mt < 4; ++mt) {
      #pragma unroll
      for (int v = 0; v < 8; ++v) {
        int gm = bm + waveM * 64 + mt * 16 + v + 8 * lhalf;
        float val = acc[mt][nt][v] + bz;
        if (do_gelu) val = gelu_tanh(val);
        size_t o = (size_t)gm * N + gn;
        if (outF) outF[o] = val;
        if (outB) outB[o] = (__bf16)val;
      }
    }
  }
}

// =====================================================================
// Flash attention: grid (qtiles=9, H, B); 8 waves x 16 queries = 128 q/block.
// qkv layout: [MP rows][1536] bf16 (q|k|v, head h at col h*64).
// =====================================================================
__global__ __launch_bounds__(256) void attn_kernel(
    const __bf16* __restrict__ qkv, const float* __restrict__ kbias,
    __bf16* __restrict__ ctx)
{
  const int qt = blockIdx.x, h = blockIdx.y, b = blockIdx.z;
  __shared__ __align__(16) __bf16 Vt[64 * GLDS];        // transposed V chunk [dk][key]
  __shared__ __align__(16) __bf16 Ps[8 * 16 * GLDS];    // per-wave P scratch

  const int tid = threadIdx.x, lane = tid & 31, wid = tid >> 5;
  const int lnn = lane & 15, lhalf = lane >> 4, kb = lhalf * 8;
  const int q0 = qt * 128 + wid * 16;
  const size_t rs = 1536;
  const __bf16* qbase = qkv + (size_t)b * N_SP * rs + (size_t)h * 64;
  const __bf16* kbse  = qbase + 512;
  const __bf16* vbse  = qbase + 1024;

  // Q fragments (A layout): lane m = lane&15, K halves split across lane halves
  int qrow = q0 + lnn; if (qrow > N_SP - 1) qrow = N_SP - 1;
  const __bf16* qp = qbase + (size_t)qrow * rs;
  Frag16 fa0, fa1;
  fa0.u[0] = *(const uint4*)(qp + kb);
  fa0.u[1] = *(const uint4*)(qp + 16 + kb);
  fa1.u[0] = *(const uint4*)(qp + 32 + kb);
  fa1.u[1] = *(const uint4*)(qp + 48 + kb);

  floatx8 acc[4];
  #pragma unroll
  for (int nt = 0; nt < 4; ++nt)
    #pragma unroll
    for (int i = 0; i < 8; ++i) acc[nt][i] = 0.0f;
  float mrow[8], lrow[8];
  #pragma unroll
  for (int v = 0; v < 8; ++v) { mrow[v] = -1e30f; lrow[v] = 0.0f; }
  const float scale = 0.125f;  // 1/sqrt(64)

  for (int kk = 0; kk < N_SP; kk += 32) {
    // prefetch next key block rows into cache (global_prefetch_b8)
    if (kk + 32 < N_SP) {
      __builtin_prefetch(kbse + (size_t)(kk + 32 + lnn) * rs, 0, 1);
      __builtin_prefetch(vbse + (size_t)(kk + 32 + (tid & 31)) * rs, 0, 1);
    }
    // stage V chunk transposed: Vt[dk][key]
    {
      int key = tid & 31, c = tid >> 5;
      const __bf16* vp = vbse + (size_t)(kk + key) * rs + c * 8;
      uint4 vv = *(const uint4*)(vp);
      const __bf16* vh = (const __bf16*)&vv;
      #pragma unroll
      for (int i = 0; i < 8; ++i) Vt[(c * 8 + i) * GLDS + key] = vh[i];
    }
    __syncthreads();

    // scores: two 16-key subtiles, each = 2 wmma over dk 0..31 / 32..63
    floatx8 s[2];
    #pragma unroll
    for (int sub = 0; sub < 2; ++sub) {
      int key = kk + sub * 16 + lnn;
      const __bf16* kp = kbse + (size_t)key * rs + kb;
      Frag16 fb0, fb1;
      fb0.u[0] = *(const uint4*)(kp);
      fb0.u[1] = *(const uint4*)(kp + 16);
      fb1.u[0] = *(const uint4*)(kp + 32);
      fb1.u[1] = *(const uint4*)(kp + 48);
      floatx8 c0;
      #pragma unroll
      for (int i = 0; i < 8; ++i) c0[i] = 0.0f;
      c0 = wmma_bf16f32(fa0.v, fb0.v, c0);
      c0 = wmma_bf16f32(fa1.v, fb1.v, c0);
      float bv = kbias[b * N_SP + key];
      #pragma unroll
      for (int i = 0; i < 8; ++i) c0[i] = c0[i] * scale + bv;
      s[sub] = c0;
    }

    // online softmax per row (rows live across the 16 lanes of each half-wave)
    float pr[2][8];
    #pragma unroll
    for (int v = 0; v < 8; ++v) {
      float mx = fmaxf(s[0][v], s[1][v]);
      #pragma unroll
      for (int off = 1; off < 16; off <<= 1) mx = fmaxf(mx, __shfl_xor(mx, off, 32));
      float nm = fmaxf(mrow[v], mx);
      float p0 = __expf(s[0][v] - nm), p1 = __expf(s[1][v] - nm);
      float rsum = p0 + p1;
      #pragma unroll
      for (int off = 1; off < 16; off <<= 1) rsum += __shfl_xor(rsum, off, 32);
      float corr = __expf(mrow[v] - nm);
      lrow[v] = lrow[v] * corr + rsum;
      mrow[v] = nm;
      #pragma unroll
      for (int nt = 0; nt < 4; ++nt) acc[nt][v] *= corr;
      pr[0][v] = p0; pr[1][v] = p1;
    }

    // D-layout f32 -> A-layout bf16 via per-wave LDS scratch
    __bf16* pw = &Ps[wid * 16 * GLDS];
    #pragma unroll
    for (int sub = 0; sub < 2; ++sub)
      #pragma unroll
      for (int v = 0; v < 8; ++v) {
        int m = v + 8 * lhalf;
        pw[m * GLDS + sub * 16 + lnn] = (__bf16)pr[sub][v];
      }
    asm volatile("s_wait_dscnt 0" ::: "memory");
    Frag16 pf;
    pf.u[0] = *(const uint4*)(&pw[lnn * GLDS + kb]);
    pf.u[1] = *(const uint4*)(&pw[lnn * GLDS + 16 + kb]);

    #pragma unroll
    for (int nt = 0; nt < 4; ++nt) {
      Frag16 vf;
      vf.u[0] = *(const uint4*)(&Vt[(nt * 16 + lnn) * GLDS + kb]);
      vf.u[1] = *(const uint4*)(&Vt[(nt * 16 + lnn) * GLDS + 16 + kb]);
      acc[nt] = wmma_bf16f32(pf.v, vf.v, acc[nt]);
    }
    __syncthreads();
  }

  #pragma unroll
  for (int nt = 0; nt < 4; ++nt)
    #pragma unroll
    for (int v = 0; v < 8; ++v) {
      int q = q0 + v + 8 * lhalf;
      if (q < N_SP) {
        float val = acc[nt][v] / lrow[v];
        ctx[(size_t)(b * N_SP + q) * N_D + h * 64 + nt * 16 + lnn] = (__bf16)val;
      }
    }
}

// =====================================================================
// residual add + LayerNorm; re-zeros padded rows.
// =====================================================================
__global__ __launch_bounds__(256) void add_ln_kernel(
    const float* __restrict__ y, float* __restrict__ x, __bf16* __restrict__ xb,
    const float* __restrict__ g, const float* __restrict__ bt)
{
  const int row = blockIdx.x;
  const int s = row % N_SP;
  const int tid = threadIdx.x;
  const size_t xo = (size_t)row * N_D;
  if (s >= N_S1) {
    #pragma unroll
    for (int e = 0; e < 2; ++e) {
      int d = tid + e * 256;
      x[xo + d] = 0.0f; xb[xo + d] = (__bf16)0.0f;
    }
    return;
  }
  __shared__ float red[2][8];
  float vals[2], sum = 0.0f, sq = 0.0f;
  #pragma unroll
  for (int e = 0; e < 2; ++e) {
    int d = tid + e * 256;
    float v = x[xo + d] + y[xo + d];
    vals[e] = v; sum += v; sq += v * v;
  }
  #pragma unroll
  for (int off = 1; off < 32; off <<= 1) {
    sum += __shfl_xor(sum, off, 32);
    sq  += __shfl_xor(sq,  off, 32);
  }
  int lane = tid & 31, wid = tid >> 5;
  if (lane == 0) { red[0][wid] = sum; red[1][wid] = sq; }
  __syncthreads();
  float ts = 0.0f, tq = 0.0f;
  #pragma unroll
  for (int w = 0; w < 8; ++w) { ts += red[0][w]; tq += red[1][w]; }
  float mean = ts * (1.0f / N_D);
  float var  = tq * (1.0f / N_D) - mean * mean;
  float rstd = rsqrtf(var + 1e-6f);
  #pragma unroll
  for (int e = 0; e < 2; ++e) {
    int d = tid + e * 256;
    float o = (vals[e] - mean) * rstd * g[d] + bt[d];
    x[xo + d] = o; xb[xo + d] = (__bf16)o;
  }
}

// copy padded x -> dense [B, S1, D] output
__global__ void copyout_kernel(const float* __restrict__ x, float* __restrict__ out) {
  long idx = (long)blockIdx.x * 256 + threadIdx.x;
  if (idx >= (long)N_B * N_S1 * N_D) return;
  int d = (int)(idx % N_D);
  long r = idx / N_D;
  int b = (int)(r / N_S1);
  int s = (int)(r % N_S1);
  out[idx] = x[((size_t)b * N_SP + s) * N_D + d];
}

// =====================================================================
// host launcher
// =====================================================================
extern "C" void kernel_launch(void* const* d_in, const int* in_sizes, int n_in,
                              void* d_out, int out_size, void* d_ws, size_t ws_size,
                              hipStream_t stream) {
  (void)in_sizes; (void)n_in; (void)out_size; (void)ws_size;
  const float* in_emb  = (const float*)d_in[0];
  const int*   in_msk  = (const int*)d_in[1];
  const float* in_rnd  = (const float*)d_in[2];
  const int*   in_perm = (const int*)d_in[3];
  const float* in_pos  = (const float*)d_in[4];
  const float* in_posg = (const float*)d_in[5];
  const float* in_posb = (const float*)d_in[6];
  const float* in_mtok = (const float*)d_in[7];
  const float* in_atok = (const float*)d_in[8];
  const float* in_wq   = (const float*)d_in[9];
  const float* in_bq   = (const float*)d_in[10];
  const float* in_wk   = (const float*)d_in[11];
  const float* in_bk   = (const float*)d_in[12];
  const float* in_wv   = (const float*)d_in[13];
  const float* in_bv   = (const float*)d_in[14];
  const float* in_wo   = (const float*)d_in[15];
  const float* in_bo   = (const float*)d_in[16];
  const float* in_l1g  = (const float*)d_in[17];
  const float* in_l1b  = (const float*)d_in[18];
  const float* in_w1   = (const float*)d_in[19];
  const float* in_b1   = (const float*)d_in[20];
  const float* in_w2   = (const float*)d_in[21];
  const float* in_b2   = (const float*)d_in[22];
  const float* in_l2g  = (const float*)d_in[23];
  const float* in_l2b  = (const float*)d_in[24];

  char* ws = (char*)d_ws;
  // 256B-aligned workspace layout
  constexpr size_t off_x    = 0;                                  // f32 [MP,512]
  constexpr size_t off_xb   = off_x    + (size_t)N_MP * 512 * 4;  // bf16 [MP,512]
  constexpr size_t off_qkv  = off_xb   + (size_t)N_MP * 512 * 2;  // bf16 [MP,1536]
  constexpr size_t off_ctx  = off_qkv  + (size_t)N_MP * 1536 * 2; // bf16 [MP,512]
  constexpr size_t off_ffn  = off_ctx  + (size_t)N_MP * 512 * 2;  // bf16 [MP,2048]
  constexpr size_t off_tmp  = off_ffn  + (size_t)N_MP * 2048 * 2; // f32 [MP,512]
  constexpr size_t off_kb   = off_tmp  + (size_t)N_MP * 512 * 4;  // f32 [B,SP]
  constexpr size_t off_bqkv = off_kb   + (size_t)N_B * N_SP * 4;  // f32 [L,1536]
  constexpr size_t off_wqkv = off_bqkv + (size_t)N_L * 1536 * 4;  // bf16 [L,1536,512] (transposed)
  constexpr size_t off_wo   = off_wqkv + (size_t)N_L * 1536 * 512 * 2; // bf16 [L,512,512] (t)
  constexpr size_t off_w1   = off_wo   + (size_t)N_L * 512 * 512 * 2;  // bf16 [L,2048,512] (t)
  constexpr size_t off_w2   = off_w1   + (size_t)N_L * 2048 * 512 * 2; // bf16 [L,512,2048] (t)

  float*  xf   = (float*)(ws + off_x);
  __bf16* xb   = (__bf16*)(ws + off_xb);
  __bf16* qkv  = (__bf16*)(ws + off_qkv);
  __bf16* ctx  = (__bf16*)(ws + off_ctx);
  __bf16* ffn  = (__bf16*)(ws + off_ffn);
  float*  tmp  = (float*)(ws + off_tmp);
  float*  kbia = (float*)(ws + off_kb);
  float*  bqkv = (float*)(ws + off_bqkv);
  __bf16* wqkv = (__bf16*)(ws + off_wqkv);
  __bf16* wo   = (__bf16*)(ws + off_wo);
  __bf16* w1   = (__bf16*)(ws + off_w1);
  __bf16* w2   = (__bf16*)(ws + off_w2);

  float* out_x    = (float*)d_out;
  float* out_emb  = out_x + (size_t)N_B * N_S1 * N_D;
  float* out_mask = out_emb + (size_t)N_B * N_S * N_D;

  // --- weight/bias conversion (all transposed to [N][K]) ---
  {
    long nq = (long)N_L * 1536 * 512;
    packqkv_kernel<<<(nq + 255) / 256, 256, 0, stream>>>(in_wq, in_wk, in_wv, wqkv);
    packbqkv_kernel<<<(N_L * 1536 + 255) / 256, 256, 0, stream>>>(in_bq, in_bk, in_bv, bqkv);
    long nwo = (long)N_L * 512 * 512;
    cvt_t_kernel<<<(nwo + 255) / 256, 256, 0, stream>>>(in_wo, wo, 512, 512, nwo);
    long nw1 = (long)N_L * 512 * 2048;
    cvt_t_kernel<<<(nw1 + 255) / 256, 256, 0, stream>>>(in_w1, w1, 512, 2048, nw1);
    cvt_t_kernel<<<(nw1 + 255) / 256, 256, 0, stream>>>(in_w2, w2, 2048, 512, nw1);
  }

  // --- embedding + bias ---
  embed_kernel<<<N_MP, 256, 0, stream>>>(in_emb, in_rnd, in_perm, in_pos,
                                         in_posg, in_posb, in_mtok, in_atok,
                                         xf, xb, out_emb, out_mask);
  biasattn_kernel<<<(N_B * N_SP + 255) / 256, 256, 0, stream>>>(in_msk, kbia);

  // --- encoder layers ---
  const int MT = N_MP / 256;  // 33
  for (int l = 0; l < N_L; ++l) {
    // fused QKV projection
    gemm_kernel<<<dim3(1536 / 128, MT), 256, 0, stream>>>(
        xb, wqkv + (size_t)l * 1536 * 512, bqkv + (size_t)l * 1536,
        nullptr, qkv, N_MP, 1536, 512, 0);
    // attention
    attn_kernel<<<dim3((N_SP + 127) / 128, N_H, N_B), 256, 0, stream>>>(qkv, kbia, ctx);
    // output projection
    gemm_kernel<<<dim3(512 / 128, MT), 256, 0, stream>>>(
        ctx, wo + (size_t)l * 512 * 512, in_bo + (size_t)l * 512,
        tmp, nullptr, N_MP, 512, 512, 0);
    add_ln_kernel<<<N_MP, 256, 0, stream>>>(tmp, xf, xb,
        in_l1g + (size_t)l * 512, in_l1b + (size_t)l * 512);
    // FFN
    gemm_kernel<<<dim3(2048 / 128, MT), 256, 0, stream>>>(
        xb, w1 + (size_t)l * 2048 * 512, in_b1 + (size_t)l * 2048,
        nullptr, ffn, N_MP, 2048, 512, 1);
    gemm_kernel<<<dim3(512 / 128, MT), 256, 0, stream>>>(
        ffn, w2 + (size_t)l * 512 * 2048, in_b2 + (size_t)l * 512,
        tmp, nullptr, N_MP, 512, 2048, 0);
    add_ln_kernel<<<N_MP, 256, 0, stream>>>(tmp, xf, xb,
        in_l2g + (size_t)l * 512, in_l2b + (size_t)l * 512);
  }

  // --- unpad to output ---
  long nout = (long)N_B * N_S1 * N_D;
  copyout_kernel<<<(nout + 255) / 256, 256, 0, stream>>>(xf, out_x);
}
